// SimpleGraphConv_18159121728118
// MI455X (gfx1250) — matrix operand order repaired and checked by reading, not compiled, not run on hardware
//
#include <hip/hip_runtime.h>

typedef __attribute__((ext_vector_type(16))) _Float16 v16h;
typedef __attribute__((ext_vector_type(8)))  float    v8f;

#define N_NODES  50000
#define N_EDGES  800000
#define HID      128
#define DEPTH    10
#define N_GRAPHS 512
#define IN_F     32
#define CONC     (DEPTH * HID)   /* 1280 */
#define SPAD     136             /* LDS row stride in f16 (128 + 8 pad) */

// ---------------------------------------------------------------------------
// Pack [DEPTH][128][128] row-major f32 weights (W_rel, W_root) into the wave32
// WMMA B-fragment f16 layout: one contiguous v16h per (kb, nt, lane):
//   packed[(d*2+m)*16384 + kb*4096 + nt*512 + lane*16 + t]
//     = W[d][kb*32 + 8*(lane>>4) + (t&7) + 16*(t>>3)][nt*16 + (lane&15)]
// ---------------------------------------------------------------------------
__global__ void pack_w_kernel(const float* __restrict__ W_rel,
                              const float* __restrict__ W_root,
                              _Float16* __restrict__ packed) {
  int tid = blockIdx.x * blockDim.x + threadIdx.x;   // 327680 total
  int t    =  tid        & 15;
  int lane = (tid >> 4)  & 31;
  int nt   = (tid >> 9)  & 7;
  int kb   = (tid >> 12) & 3;
  int m    = (tid >> 14) & 1;
  int d    =  tid >> 15;
  if (d >= DEPTH) return;
  int k = kb * 32 + ((lane >> 4) << 3) + (t & 7) + ((t >> 3) << 4);
  int n = nt * 16 + (lane & 15);
  const float* W = (m == 0) ? W_rel : W_root;
  float v = W[(size_t)d * HID * HID + (size_t)k * HID + n];
  size_t o = (size_t)(d * 2 + m) * (HID * HID)
           + (size_t)kb * 4096 + (size_t)nt * 512 + (size_t)lane * 16 + t;
  packed[o] = (_Float16)v;
}

// ---------------------------------------------------------------------------
// h0 = f16( x @ W_embed )   ([50000,32] @ [32,128])
// ---------------------------------------------------------------------------
__global__ void embed_kernel(const float* __restrict__ x,
                             const float* __restrict__ W_embed,
                             _Float16* __restrict__ h0) {
  int n = blockIdx.x;          // node
  int c = threadIdx.x;         // out feature
  const float* xr = x + (size_t)n * IN_F;
  float acc = 0.f;
  #pragma unroll
  for (int k = 0; k < IN_F; ++k)
    acc = fmaf(xr[k], W_embed[(size_t)k * HID + c], acc);
  h0[(size_t)n * HID + c] = (_Float16)acc;
}

__global__ void zero_kernel(float* __restrict__ p, int n) {
  int i = blockIdx.x * blockDim.x + threadIdx.x;
  if (i < n) p[i] = 0.f;
}

// ---------------------------------------------------------------------------
// agg[dst[e]] += h_prev[src[e]]  (f16 gather, f32 atomic accumulate).
// h (12.8 MB) and agg (25.6 MB) are L2-resident -> atomics stay on-chip.
// ---------------------------------------------------------------------------
__global__ void scatter_kernel(const _Float16* __restrict__ h_prev,
                               const int* __restrict__ eidx,
                               float* __restrict__ agg) {
  size_t tid = (size_t)blockIdx.x * blockDim.x + threadIdx.x;
  int e = (int)(tid >> 7);
  int c = (int)(tid & 127);
  if (e >= N_EDGES) return;
  int s = eidx[e];
  int d = eidx[N_EDGES + e];
  atomicAdd(&agg[(size_t)d * HID + c], (float)h_prev[(size_t)s * HID + c]);
}

// ---------------------------------------------------------------------------
// Build a 16x32 WMMA A-fragment from an LDS-staged [16][SPAD] f16 tile.
// CDNA5 16-bit A layout: lane<16 -> row lane, K {0..7,16..23} of the block;
// lane>=16 -> row lane-16, K {8..15,24..31}.  Two 16B ds loads per lane.
// ---------------------------------------------------------------------------
__device__ __forceinline__ v16h frag_from_lds(const _Float16* __restrict__ s,
                                              int kb, int lane) {
  int l  = lane & 15;
  int hi = lane >> 4;
  const _Float16* p = s + l * SPAD + kb * 32 + hi * 8;
  v16h a;
  #pragma unroll
  for (int j = 0; j < 8; ++j) {
    a[j]     = p[j];
    a[8 + j] = p[16 + j];
  }
  return a;
}

// ---------------------------------------------------------------------------
// h_next = f16( agg @ W_rel[d] + b_rel[d] + h_prev @ W_root[d] )
// Fused epilogue: pooled[batch[m], d*128 + n] += value (f32).
// Block = 256 thr = 8 waves sharing one 16-row M-tile staged in LDS;
// wave nt owns N-tile nt.  50000/16 = 3125 exact -> EXEC all-1s for WMMA.
// ---------------------------------------------------------------------------
__global__ __launch_bounds__(256)
void depth_gemm_kernel(const _Float16* __restrict__ h_prev,
                       const float* __restrict__ agg,
                       const _Float16* __restrict__ packed,
                       int depth,
                       const float* __restrict__ b_rel,
                       const int* __restrict__ batch,
                       _Float16* __restrict__ h_next,
                       float* __restrict__ pooled) {
  __shared__ _Float16 sAgg[16 * SPAD];
  __shared__ _Float16 sH[16 * SPAD];
  const int tid = threadIdx.x;
  const int m0  = blockIdx.x * 16;

  // Cooperative stage: each thread covers 8 columns of one row.
  {
    int r  = tid >> 4;
    int c0 = (tid & 15) * 8;
    const float* ap = agg + (size_t)(m0 + r) * HID + c0;
    _Float16* da = &sAgg[r * SPAD + c0];
    #pragma unroll
    for (int j = 0; j < 8; ++j) da[j] = (_Float16)ap[j];   // convert once/block
    const _Float16* hp = h_prev + (size_t)(m0 + r) * HID + c0;
    _Float16* dh = &sH[r * SPAD + c0];
    #pragma unroll
    for (int j = 0; j < 8; ++j) dh[j] = hp[j];
  }
  __syncthreads();

  const int lane = tid & 31;
  const int nt   = tid >> 5;                 // 0..7 -> N-tile
  const _Float16* Wrel  = packed + (size_t)(depth * 2 + 0) * (HID * HID);
  const _Float16* Wroot = packed + (size_t)(depth * 2 + 1) * (HID * HID);

  v8f c = {};
  #pragma unroll
  for (int kb = 0; kb < 4; ++kb) {
    v16h a = frag_from_lds(sAgg, kb, lane);
    v16h b = *(const v16h*)(Wrel + (size_t)kb * 4096 + (size_t)nt * 512 + (size_t)lane * 16);
    c = __builtin_amdgcn_wmma_f32_16x16x32_f16(false, a, false, b,
                                               (short)0, c, false, false);
  }
  #pragma unroll
  for (int kb = 0; kb < 4; ++kb) {
    v16h a = frag_from_lds(sH, kb, lane);
    v16h b = *(const v16h*)(Wroot + (size_t)kb * 4096 + (size_t)nt * 512 + (size_t)lane * 16);
    c = __builtin_amdgcn_wmma_f32_16x16x32_f16(false, a, false, b,
                                               (short)0, c, false, false);
  }

  // C/D layout: VGPR r, lanes 0-15 -> row m0+r, lanes 16-31 -> row m0+8+r;
  // column n = nt*16 + lane%16.
  const int l  = lane & 15;
  const int hi = lane >> 4;
  const int n  = nt * 16 + l;
  const float bias = b_rel[depth * HID + n];
  #pragma unroll
  for (int r = 0; r < 8; ++r) {
    int m = m0 + r + 8 * hi;
    float v = c[r] + bias;
    h_next[(size_t)m * HID + n] = (_Float16)v;
    int g = batch[m];
    atomicAdd(&pooled[(size_t)g * CONC + depth * HID + n], v);
  }
}

// ---------------------------------------------------------------------------
// out = pooled @ W_sg + b_sg   ([512,1280] @ [1280,32])
// ---------------------------------------------------------------------------
__global__ void final_gemm_kernel(const float* __restrict__ pooled,
                                  const float* __restrict__ W_sg,
                                  const float* __restrict__ b_sg,
                                  float* __restrict__ out) {
  int tid = blockIdx.x * blockDim.x + threadIdx.x;   // 16384 total
  int g = tid >> 5;
  int o = tid & 31;
  if (g >= N_GRAPHS) return;
  const float* pr = pooled + (size_t)g * CONC;
  float acc = b_sg[o];
  for (int k = 0; k < CONC; ++k)
    acc = fmaf(pr[k], W_sg[(size_t)k * IN_F + o], acc);
  out[(size_t)g * IN_F + o] = acc;
}

// ---------------------------------------------------------------------------
extern "C" void kernel_launch(void* const* d_in, const int* in_sizes, int n_in,
                              void* d_out, int out_size, void* d_ws, size_t ws_size,
                              hipStream_t stream) {
  (void)in_sizes; (void)n_in; (void)out_size; (void)ws_size;
  const float* x       = (const float*)d_in[0];
  const int*   eidx    = (const int*)  d_in[1];
  const int*   batch   = (const int*)  d_in[2];
  /* d_in[3] = num_graphs (==512, compile-time constant here)            */
  const float* W_embed = (const float*)d_in[4];
  const float* W_rel   = (const float*)d_in[5];
  const float* b_rel   = (const float*)d_in[6];
  const float* W_root  = (const float*)d_in[7];
  /* d_in[8]=W_ws, d_in[9]=b_ws: softmax over a length-1 axis == 1 -> dead */
  const float* W_sg    = (const float*)d_in[10];
  const float* b_sg    = (const float*)d_in[11];
  float* out = (float*)d_out;

  // workspace carve-up (~55 MB total)
  char* ws = (char*)d_ws;
  size_t off = 0;
  auto take = [&](size_t bytes) -> void* {
    void* p = (void*)(ws + off);
    off += (bytes + 255) & ~(size_t)255;
    return p;
  };
  _Float16* h_a    = (_Float16*)take(sizeof(_Float16) * (size_t)N_NODES * HID);
  _Float16* h_b    = (_Float16*)take(sizeof(_Float16) * (size_t)N_NODES * HID);
  float*    agg    = (float*)   take(sizeof(float)    * (size_t)N_NODES * HID);
  float*    pooled = (float*)   take(sizeof(float)    * (size_t)N_GRAPHS * CONC);
  _Float16* packed = (_Float16*)take(sizeof(_Float16) * (size_t)DEPTH * 2 * HID * HID);

  // one-time-per-launch prep (deterministic, capture-safe)
  pack_w_kernel<<<(DEPTH * 2 * HID * HID + 255) / 256, 256, 0, stream>>>(W_rel, W_root, packed);
  embed_kernel<<<N_NODES, HID, 0, stream>>>(x, W_embed, h_a);
  zero_kernel<<<(N_GRAPHS * CONC + 255) / 256, 256, 0, stream>>>(pooled, N_GRAPHS * CONC);

  _Float16* hp = h_a;
  _Float16* hn = h_b;
  for (int d = 0; d < DEPTH; ++d) {
    zero_kernel<<<(N_NODES * HID + 255) / 256, 256, 0, stream>>>(agg, N_NODES * HID);
    scatter_kernel<<<(int)(((size_t)N_EDGES * HID) / 256), 256, 0, stream>>>(hp, eidx, agg);
    depth_gemm_kernel<<<N_NODES / 16, 256, 0, stream>>>(hp, agg, packed, d, b_rel,
                                                        batch, hn, pooled);
    _Float16* t = hp; hp = hn; hn = t;
  }

  final_gemm_kernel<<<(N_GRAPHS * IN_F + 255) / 256, 256, 0, stream>>>(pooled, W_sg, b_sg, out);
}
